// MyGraphTransformer_26242250179023
// MI455X (gfx1250) — matrix operand order repaired and checked by reading, not compiled, not run on hardware
//
#include <hip/hip_runtime.h>
#include <hip/hip_bf16.h>
#include <math.h>

typedef __attribute__((ext_vector_type(16))) _Float16 v16h;
typedef __attribute__((ext_vector_type(8)))  _Float16 v8h;
typedef __attribute__((ext_vector_type(8)))  float    v8f;

#define NTOT 4900
#define NP   4904      // padded f16 row stride (multiple of 8 halves = 16B)
#define N1G  708
#define INF_ 512
#define OUTF 256
#define HD   64

// ---------------------------------------------------------------------------
// Tiled WMMA GEMM (NT form only):  C[M,N] = scale * (A @ Bt^T)  (+ bias[N])
//   A  : f16 [M,K] row-major, lda  (lda % 8 == 0, 16B-aligned base)
//   Bt : f16 [N,K] row-major, ldb  (ldb % 8 == 0) -- i.e. B pre-transposed
//   C  : f32 (OUTF16=0) or f16 (OUTF16=1), row-major ldc
// 256 threads = 8 wave32 waves (4 along M x 2 along N).
// Block tile BM=128 x BN(template) x BK=32; wave tile 32 x BN/2
//   -> 2 x (BN/32) v_wmma_f32_16x16x32_f16 per wave per K-step.
// Double-buffered LDS; register-staged b128 global loads overlap the WMMAs;
// branchless clamped loads on edge tiles (no exec-mask juggling).
// ---------------------------------------------------------------------------
template<int BN, bool OUTF16, bool BIAS>
__global__ __launch_bounds__(256)
void gemm_wmma(const _Float16* __restrict__ A, const _Float16* __restrict__ Bt,
               void* __restrict__ C, const float* __restrict__ bias,
               int M, int N, int K, int lda, int ldb, int ldc, float scale)
{
    constexpr int BM = 128, BK = 32, LA = 40, LB = 40;
    constexpr int NF = BN / 32;              // n-frags per wave
    constexpr int BH = (BN * BK) / 256;      // B halves staged per thread (8|16)
    __shared__ __align__(16) _Float16 As[2 * BM * LA];
    __shared__ __align__(16) _Float16 Bs[2 * BN * LB];

    const int t    = threadIdx.x;
    const int lane = t & 31;
    const int wid  = t >> 5;
    const int wm   = (wid >> 1) * 32;
    const int wn   = (wid & 1) * (BN / 2);
    const bool hi  = lane >= 16;
    const int l15  = lane & 15;
    const int mbase = blockIdx.y * BM;
    const int nbase = blockIdx.x * BN;
    const bool intMN = (mbase + BM <= M) && (nbase + BN <= N);

    // staging coordinates (A: 2 thr/row, 16 halves; B: BH halves/thread)
    const int arow  = t >> 1, ahoff = (t & 1) * 16;
    const int brow  = (BH == 16) ? (t >> 1) : (t >> 2);
    const int bkoff = (BH == 16) ? ((t & 1) * 16) : ((t & 3) * 8);

    union { v8h v[2]; _Float16 h[16]; } areg, breg;

    auto gload = [&](int k0) {
        if (intMN && (k0 + BK <= K)) {                 // b128 fast path
            const v8h* ap = (const v8h*)(A + (size_t)(mbase + arow) * lda + k0 + ahoff);
            areg.v[0] = ap[0];
            areg.v[1] = ap[1];
            const v8h* bp = (const v8h*)(Bt + (size_t)(nbase + brow) * ldb + k0 + bkoff);
            breg.v[0] = bp[0];
            if constexpr (BH == 16) breg.v[1] = bp[1];
        } else {                                       // branchless clamped path
            int gm = mbase + arow;
#pragma unroll
            for (int j = 0; j < 16; ++j) {
                int gk = k0 + ahoff + j;
                bool ok = (gm < M) && (gk < K);
                _Float16 v = A[ok ? ((size_t)gm * lda + gk) : (size_t)0];
                areg.h[j] = ok ? v : (_Float16)0.f;
            }
            int gn = nbase + brow;
#pragma unroll
            for (int j = 0; j < BH; ++j) {
                int gk = k0 + bkoff + j;
                bool ok = (gn < N) && (gk < K);
                _Float16 v = Bt[ok ? ((size_t)gn * ldb + gk) : (size_t)0];
                breg.h[j] = ok ? v : (_Float16)0.f;
            }
        }
    };

    auto lstore = [&](int buf) {
        _Float16* as = As + buf * (BM * LA);
        _Float16* bs = Bs + buf * (BN * LB);
        *(v8h*)&as[arow * LA + ahoff]     = areg.v[0];
        *(v8h*)&as[arow * LA + ahoff + 8] = areg.v[1];
        *(v8h*)&bs[brow * LB + bkoff] = breg.v[0];
        if constexpr (BH == 16) *(v8h*)&bs[brow * LB + bkoff + 8] = breg.v[1];
    };

    v8f acc[2][NF];
#pragma unroll
    for (int i = 0; i < 2; ++i)
#pragma unroll
        for (int j = 0; j < NF; ++j)
#pragma unroll
            for (int v = 0; v < 8; ++v) acc[i][j][v] = 0.f;

    const int nsteps = (K + BK - 1) / BK;
    gload(0);
    lstore(0);
    __syncthreads();

    for (int i = 0; i < nsteps; ++i) {
        if (i + 1 < nsteps) gload((i + 1) * BK);       // issue next tile's loads

        const _Float16* as = As + (i & 1) * (BM * LA);
        const _Float16* bs = Bs + (i & 1) * (BN * LB);
        union { v16h v; v8h h[2]; } af[2], bf[NF];
#pragma unroll
        for (int f = 0; f < 2; ++f) {
            int mr = (wm + f * 16 + l15) * LA;
            af[f].h[0] = *(const v8h*)&as[mr + (hi ? 8 : 0)];    // K 0..7 / 8..15
            af[f].h[1] = *(const v8h*)&as[mr + (hi ? 24 : 16)];  // K 16..23 / 24..31
        }
#pragma unroll
        for (int g = 0; g < NF; ++g) {
            int nr = (wn + g * 16 + l15) * LB;
            bf[g].h[0] = *(const v8h*)&bs[nr + (hi ? 8 : 0)];    // K 0..7 / 8..15
            bf[g].h[1] = *(const v8h*)&bs[nr + (hi ? 24 : 16)];  // K 16..23 / 24..31
        }
#pragma unroll
        for (int f = 0; f < 2; ++f)
#pragma unroll
            for (int g = 0; g < NF; ++g)
                acc[f][g] = __builtin_amdgcn_wmma_f32_16x16x32_f16(
                    false, af[f].v, false, bf[g].v, (short)0, acc[f][g], false, false);

        if (i + 1 < nsteps) lstore((i + 1) & 1);       // waits loads, fills other buf
        __syncthreads();
    }

    // ---- store C (M = vgpr + (lane>=16?8:0), N = lane&15) ----
#pragma unroll
    for (int f = 0; f < 2; ++f)
#pragma unroll
        for (int g = 0; g < NF; ++g)
#pragma unroll
            for (int v = 0; v < 8; ++v) {
                int gm = mbase + wm + f * 16 + v + (hi ? 8 : 0);
                int gn = nbase + wn + g * 16 + l15;
                if (gm < M && gn < N) {
                    float val = acc[f][g][v] * scale;
                    if (BIAS) val += bias[gn];
                    if (OUTF16) ((_Float16*)C)[(size_t)gm * ldc + gn] = (_Float16)val;
                    else        ((float*)C)[(size_t)gm * ldc + gn] = val;
                }
            }
}

// ---------------------------------------------------------------------------
// Helper kernels
// ---------------------------------------------------------------------------
__global__ void k_cvt_f16(const float* __restrict__ src, _Float16* __restrict__ dst, size_t n) {
    size_t i = (size_t)blockIdx.x * blockDim.x + threadIdx.x;
    if (i < n) dst[i] = (_Float16)src[i];
}

// f32 [K,N] -> transposed f16 [N,K] (small weight matrices)
__global__ void k_cvtT(const float* __restrict__ src, _Float16* __restrict__ dst,
                       int K, int N) {
    int idx = blockIdx.x * blockDim.x + threadIdx.x;
    if (idx >= K * N) return;
    int k = idx % K, n = idx / K;
    dst[(size_t)n * K + k] = (_Float16)src[(size_t)k * N + n];
}

// f32 [NTOT,NTOT] -> transposed f16 [NTOT, NP] via LDS-tiled transpose
__global__ __launch_bounds__(256)
void k_transpose_cvt(const float* __restrict__ src, _Float16* __restrict__ dst) {
    __shared__ float tile[16][17];
    int tx = threadIdx.x & 15, ty = threadIdx.x >> 4;
    int rb = blockIdx.y * 16, cb = blockIdx.x * 16;
    int r = rb + ty, c = cb + tx;
    tile[ty][tx] = (r < NTOT && c < NTOT) ? src[(size_t)r * NTOT + c] : 0.f;
    __syncthreads();
    int ro = cb + ty, co = rb + tx;
    if (ro < NTOT && co < NTOT) dst[(size_t)ro * NP + co] = (_Float16)tile[tx][ty];
}

// rp = r @ Wr + br (4x256 @ 256x256, tiny -> scalar)
__global__ __launch_bounds__(256)
void k_rp(const float* __restrict__ r, const float* __restrict__ Wr,
          const float* __restrict__ br, float* __restrict__ rp) {
    int row = blockIdx.x, j = threadIdx.x;
    float acc = br[j];
    for (int k = 0; k < 256; ++k) acc += r[row * 256 + k] * Wr[k * 256 + j];
    rp[row * 256 + j] = acc;
}

// relation-modulated per-head Q/K f16 arrays + transposed V, reproducing the
// source's raw reshape: (h,n,d) -> row=(h*nrows+n)>>2, col=64*((h*nrows+n)&3)+d
__global__ void k_modulate(const float* __restrict__ hQ, const float* __restrict__ hK,
                           const float* __restrict__ hV, const float* __restrict__ rp,
                           const int* __restrict__ flag,
                           _Float16* QA, _Float16* QB, _Float16* QC,
                           _Float16* KA, _Float16* KB, _Float16* KC,
                           _Float16* V16T)
{
    int idx = blockIdx.x * blockDim.x + threadIdx.x;
    const int total = 4 * NTOT * HD;
    if (idx >= total) return;
    int d = idx & (HD - 1);
    int n = (idx >> 6) % NTOT;
    int h = idx / (HD * NTOT);
    int ridx = (*flag) ? 1 : 0;

    { // V transposed: V16T[h][d][n], padded K-stride NP
        int f = h * NTOT + n; int row = f >> 2; int col = ((f & 3) << 6) + d;
        V16T[(size_t)h * HD * NP + (size_t)d * NP + n] =
            (_Float16)hV[(size_t)row * OUTF + col];
    }
    if (n < N1G) {
        int f = h * N1G + n; int row = f >> 2; int col = ((f & 3) << 6) + d;
        float q  = hQ[(size_t)row * OUTF + col];
        float kk = hK[(size_t)row * OUTF + col];
        QA[idx] = (_Float16)(q  * rp[ridx * OUTF + col]);
        QB[idx] = (_Float16)(q  * rp[2    * OUTF + col]);
        QC[idx] = (_Float16)0.f;
        KA[idx] = (_Float16)(kk * rp[ridx * OUTF + col]);
        KB[idx] = (_Float16)(kk * rp[2    * OUTF + col]);
        KC[idx] = (_Float16)0.f;
    } else {
        int n2 = n - N1G;
        int f = h * (NTOT - N1G) + n2;
        int row = N1G + (f >> 2); int col = ((f & 3) << 6) + d;
        float q  = hQ[(size_t)row * OUTF + col];
        float kk = hK[(size_t)row * OUTF + col];
        QB[idx] = (_Float16)(q  * rp[2 * OUTF + col]);
        QC[idx] = (_Float16)(q  * rp[3 * OUTF + col]);
        QA[idx] = (_Float16)0.f;
        KB[idx] = (_Float16)(kk * rp[2 * OUTF + col]);
        KC[idx] = (_Float16)(kk * rp[3 * OUTF + col]);
        KA[idx] = (_Float16)0.f;
    }
}

// softmax over head axis (dim=0 of [H,N,N]); 2D grid, padded stride
__global__ void k_softmax_heads(_Float16* __restrict__ G) {
    int k = blockIdx.x * blockDim.x + threadIdx.x;
    int n = blockIdx.y;
    if (k >= NTOT) return;
    const size_t HS = (size_t)NP * NTOT;
    size_t idx = (size_t)n * NP + k;
    float x0 = (float)G[idx],          x1 = (float)G[HS + idx];
    float x2 = (float)G[2 * HS + idx], x3 = (float)G[3 * HS + idx];
    float m = fmaxf(fmaxf(x0, x1), fmaxf(x2, x3));
    float e0 = __expf(x0 - m), e1 = __expf(x1 - m);
    float e2 = __expf(x2 - m), e3 = __expf(x3 - m);
    float rs = 1.f / (e0 + e1 + e2 + e3);
    G[idx]          = (_Float16)(e0 * rs);
    G[HS + idx]     = (_Float16)(e1 * rs);
    G[2 * HS + idx] = (_Float16)(e2 * rs);
    G[3 * HS + idx] = (_Float16)(e3 * rs);
}

__global__ void k_pack_H(const float* __restrict__ H, _Float16* __restrict__ cat16) {
    int idx = blockIdx.x * blockDim.x + threadIdx.x;
    if (idx >= NTOT * OUTF) return;
    int n = idx >> 8, j = idx & 255;
    cat16[(size_t)n * 512 + j] = (_Float16)H[idx];
}

// gather heads, relu, LayerNorm(1e-5); writes f32 mln and f16 into cat[:,256:]
__global__ __launch_bounds__(256)
void k_mrelu_ln(const float* __restrict__ mh, const float* __restrict__ g,
                const float* __restrict__ b, float* __restrict__ mln,
                _Float16* __restrict__ cat16) {
    __shared__ float s1[256], s2[256];
    int n = blockIdx.x, t = threadIdx.x;
    int h = t >> 6, d = t & 63;
    float x = mh[(size_t)h * NTOT * HD + (size_t)n * HD + d];
    x = fmaxf(x, 0.f);
    s1[t] = x; s2[t] = x * x;
    __syncthreads();
    for (int s = 128; s > 0; s >>= 1) {
        if (t < s) { s1[t] += s1[t + s]; s2[t] += s2[t + s]; }
        __syncthreads();
    }
    float mu  = s1[0] * (1.f / 256.f);
    float var = s2[0] * (1.f / 256.f) - mu * mu;
    float y = (x - mu) * rsqrtf(var + 1e-5f) * g[t] + b[t];
    mln[(size_t)n * OUTF + t] = y;
    cat16[(size_t)n * 512 + OUTF + t] = (_Float16)y;
}

__global__ void k_gate(const float* __restrict__ bout, const float* __restrict__ mln,
                       const float* __restrict__ H, _Float16* __restrict__ out16) {
    int idx = blockIdx.x * blockDim.x + threadIdx.x;
    if (idx >= NTOT * OUTF) return;
    float beta = 1.f / (1.f + __expf(-bout[idx]));
    out16[idx] = (_Float16)(beta * mln[idx] + (1.f - beta) * H[idx]);
}

__global__ __launch_bounds__(256)
void k_ln_tanh(const float* __restrict__ tin, const float* __restrict__ g,
               const float* __restrict__ b, _Float16* __restrict__ h16) {
    __shared__ float s1[256], s2[256];
    int n = blockIdx.x, t = threadIdx.x;
    float x = tin[(size_t)n * OUTF + t];
    s1[t] = x; s2[t] = x * x;
    __syncthreads();
    for (int s = 128; s > 0; s >>= 1) {
        if (t < s) { s1[t] += s1[t + s]; s2[t] += s2[t + s]; }
        __syncthreads();
    }
    float mu  = s1[0] * (1.f / 256.f);
    float var = s2[0] * (1.f / 256.f) - mu * mu;
    float y = tanhf((x - mu) * rsqrtf(var + 1e-6f) * g[t] + b[t]);
    h16[(size_t)n * OUTF + t] = (_Float16)y;
}

// ---------------------------------------------------------------------------
extern "C" void kernel_launch(void* const* d_in, const int* in_sizes, int n_in,
                              void* d_out, int out_size, void* d_ws, size_t ws_size,
                              hipStream_t stream) {
    const float* feature = (const float*)d_in[0];
    const float* adj     = (const float*)d_in[1];
    const float* r       = (const float*)d_in[2];
    const float* Wq = (const float*)d_in[3];   const float* bq = (const float*)d_in[4];
    const float* Wk = (const float*)d_in[5];   const float* bk = (const float*)d_in[6];
    const float* Wv = (const float*)d_in[7];   const float* bv = (const float*)d_in[8];
    const float* Ww = (const float*)d_in[9];   const float* bw = (const float*)d_in[10];
    const float* Wr = (const float*)d_in[11];  const float* br = (const float*)d_in[12];
    const float* Wbeta = (const float*)d_in[13]; const float* bbeta = (const float*)d_in[14];
    const float* ln1g = (const float*)d_in[15];  const float* ln1b = (const float*)d_in[16];
    const float* w1 = (const float*)d_in[17];    const float* b1 = (const float*)d_in[18];
    const float* ln2g = (const float*)d_in[19];  const float* ln2b = (const float*)d_in[20];
    const float* w2 = (const float*)d_in[21];
    const int* flag = (const int*)d_in[22];
    float* out = (float*)d_out;

    char* ws = (char*)d_ws;
    size_t off = 0;
    auto alloc = [&](size_t bytes) -> void* {
        void* p = ws + off;
        off = (off + bytes + 255) & ~(size_t)255;
        return p;
    };
    const size_t NNP = (size_t)NP * NTOT;   // padded plane (halves)

    _Float16* feat16  = (_Float16*)alloc((size_t)NTOT * INF_ * 2);
    _Float16* WqT     = (_Float16*)alloc((size_t)INF_ * OUTF * 2);  // [256][512]
    _Float16* WkT     = (_Float16*)alloc((size_t)INF_ * OUTF * 2);
    _Float16* WvT     = (_Float16*)alloc((size_t)INF_ * OUTF * 2);
    _Float16* WwT     = (_Float16*)alloc((size_t)INF_ * OUTF * 2);
    _Float16* WbetaT  = (_Float16*)alloc((size_t)2 * OUTF * OUTF * 2); // [256][512]
    _Float16* w1T     = (_Float16*)alloc((size_t)OUTF * OUTF * 2);
    _Float16* w2T     = (_Float16*)alloc((size_t)OUTF * OUTF * 2);
    float*    rp      = (float*)   alloc(4 * OUTF * 4);
    float*    Hbuf    = (float*)   alloc((size_t)NTOT * OUTF * 4);
    float*    hQ      = (float*)   alloc((size_t)NTOT * OUTF * 4);
    float*    hK      = (float*)   alloc((size_t)NTOT * OUTF * 4);
    float*    hV      = (float*)   alloc((size_t)NTOT * OUTF * 4);
    _Float16* QA = (_Float16*)alloc((size_t)4 * NTOT * HD * 2);
    _Float16* QB = (_Float16*)alloc((size_t)4 * NTOT * HD * 2);
    _Float16* QC = (_Float16*)alloc((size_t)4 * NTOT * HD * 2);
    _Float16* KA = (_Float16*)alloc((size_t)4 * NTOT * HD * 2);
    _Float16* KB = (_Float16*)alloc((size_t)4 * NTOT * HD * 2);
    _Float16* KC = (_Float16*)alloc((size_t)4 * NTOT * HD * 2);
    _Float16* V16T  = (_Float16*)alloc((size_t)4 * HD * NP * 2);   // [h][64][NP]
    _Float16* adjT  = (_Float16*)alloc(NNP * 2);                   // adj^T, stride NP
    _Float16* S16   = (_Float16*)alloc(NNP * 2);                   // one head, stride NP
    _Float16* Gbuf  = (_Float16*)alloc(4 * NNP * 2);               // logits->probs
    float*    mh    = (float*)   alloc((size_t)4 * NTOT * HD * 4);
    _Float16* cat16 = (_Float16*)alloc((size_t)NTOT * 512 * 2);
    float*    mln   = (float*)   alloc((size_t)NTOT * OUTF * 4);
    float*    bout  = (float*)   alloc((size_t)NTOT * OUTF * 4);
    _Float16* out16 = (_Float16*)alloc((size_t)NTOT * OUTF * 2);
    float*    tbuf  = (float*)   alloc((size_t)NTOT * OUTF * 4);
    _Float16* h16   = (_Float16*)alloc((size_t)NTOT * OUTF * 2);

    dim3 blk(256);
    auto grd  = [](int M, int N) { return dim3((N + 127) / 128, (M + 127) / 128); };
    auto grd64= [](int M, int N) { return dim3((N + 63) / 64,   (M + 127) / 128); };
    auto gcnt = [](size_t n) { return dim3((unsigned)((n + 255) / 256)); };

    // ---- conversions (weights transposed so every GEMM is A @ Bt^T) ----
    k_cvt_f16<<<gcnt((size_t)NTOT * INF_), blk, 0, stream>>>(feature, feat16, (size_t)NTOT * INF_);
    k_cvtT<<<gcnt((size_t)INF_ * OUTF), blk, 0, stream>>>(Wq, WqT, INF_, OUTF);
    k_cvtT<<<gcnt((size_t)INF_ * OUTF), blk, 0, stream>>>(Wk, WkT, INF_, OUTF);
    k_cvtT<<<gcnt((size_t)INF_ * OUTF), blk, 0, stream>>>(Wv, WvT, INF_, OUTF);
    k_cvtT<<<gcnt((size_t)INF_ * OUTF), blk, 0, stream>>>(Ww, WwT, INF_, OUTF);
    k_cvtT<<<gcnt((size_t)2 * OUTF * OUTF), blk, 0, stream>>>(Wbeta, WbetaT, 2 * OUTF, OUTF);
    k_cvtT<<<gcnt((size_t)OUTF * OUTF), blk, 0, stream>>>(w1, w1T, OUTF, OUTF);
    k_cvtT<<<gcnt((size_t)OUTF * OUTF), blk, 0, stream>>>(w2, w2T, OUTF, OUTF);
    k_transpose_cvt<<<dim3((NTOT + 15) / 16, (NTOT + 15) / 16), blk, 0, stream>>>(adj, adjT);

    // ---- projections: X @ W == X @ (W^T)^T ----
    gemm_wmma<128, false, true><<<grd(NTOT, OUTF), blk, 0, stream>>>(
        feat16, WwT, Hbuf, bw, NTOT, OUTF, INF_, INF_, INF_, OUTF, 1.f);
    gemm_wmma<128, false, true><<<grd(NTOT, OUTF), blk, 0, stream>>>(
        feat16, WqT, hQ, bq, NTOT, OUTF, INF_, INF_, INF_, OUTF, 1.f);
    gemm_wmma<128, false, true><<<grd(NTOT, OUTF), blk, 0, stream>>>(
        feat16, WkT, hK, bk, NTOT, OUTF, INF_, INF_, INF_, OUTF, 1.f);
    gemm_wmma<128, false, true><<<grd(NTOT, OUTF), blk, 0, stream>>>(
        feat16, WvT, hV, bv, NTOT, OUTF, INF_, INF_, INF_, OUTF, 1.f);

    // ---- rp, modulation ----
    k_rp<<<dim3(4), blk, 0, stream>>>(r, Wr, br, rp);
    k_modulate<<<gcnt((size_t)4 * NTOT * HD), blk, 0, stream>>>(
        hQ, hK, hV, rp, flag, QA, QB, QC, KA, KB, KC, V16T);

    // ---- per head: block-structured scores (Q @ K^T), then S @ adj ----
    for (int h = 0; h < 4; ++h) {
        const _Float16* QAh = QA + (size_t)h * NTOT * HD;
        const _Float16* QBh = QB + (size_t)h * NTOT * HD;
        const _Float16* QCh = QC + (size_t)h * NTOT * HD;
        const _Float16* KAh = KA + (size_t)h * NTOT * HD;
        const _Float16* KBh = KB + (size_t)h * NTOT * HD;
        const _Float16* KCh = KC + (size_t)h * NTOT * HD;
        gemm_wmma<128, true, false><<<grd(NTOT, NTOT), blk, 0, stream>>>(
            QBh, KBh, S16, nullptr, NTOT, NTOT, HD, HD, HD, NP, 0.5f);
        gemm_wmma<128, true, false><<<grd(N1G, N1G), blk, 0, stream>>>(
            QAh, KAh, S16, nullptr, N1G, N1G, HD, HD, HD, NP, 0.5f);
        gemm_wmma<128, true, false><<<grd(NTOT - N1G, NTOT - N1G), blk, 0, stream>>>(
            QCh + (size_t)N1G * HD, KCh + (size_t)N1G * HD,
            S16 + (size_t)N1G * NP + N1G, nullptr,
            NTOT - N1G, NTOT - N1G, HD, HD, HD, NP, 0.5f);
        // G[h] = S @ adj = S @ (adjT)^T   (4900^3 MACs)
        gemm_wmma<128, true, false><<<grd(NTOT, NTOT), blk, 0, stream>>>(
            S16, adjT, Gbuf + (size_t)h * NNP, nullptr,
            NTOT, NTOT, NTOT, NP, NP, NP, 1.f);
    }

    // ---- softmax over heads ----
    k_softmax_heads<<<dim3((NTOT + 255) / 256, NTOT), blk, 0, stream>>>(Gbuf);

    // ---- m[h] = P[h] @ V[h] = P[h] @ (V16T[h])^T  (N=64 -> BN=64 variant) ----
    for (int h = 0; h < 4; ++h) {
        gemm_wmma<64, false, false><<<grd64(NTOT, HD), blk, 0, stream>>>(
            Gbuf + (size_t)h * NNP, V16T + (size_t)h * HD * NP,
            mh + (size_t)h * NTOT * HD, nullptr,
            NTOT, HD, NTOT, NP, NP, HD, 1.f);
    }

    // ---- tail ----
    k_pack_H<<<gcnt((size_t)NTOT * OUTF), blk, 0, stream>>>(Hbuf, cat16);
    k_mrelu_ln<<<dim3(NTOT), blk, 0, stream>>>(mh, ln1g, ln1b, mln, cat16);
    gemm_wmma<128, false, true><<<grd(NTOT, OUTF), blk, 0, stream>>>(
        cat16, WbetaT, bout, bbeta, NTOT, OUTF, 2 * OUTF, 2 * OUTF, 2 * OUTF, OUTF, 1.f);
    k_gate<<<gcnt((size_t)NTOT * OUTF), blk, 0, stream>>>(bout, mln, Hbuf, out16);
    gemm_wmma<128, false, true><<<grd(NTOT, OUTF), blk, 0, stream>>>(
        out16, w1T, tbuf, b1, NTOT, OUTF, OUTF, OUTF, OUTF, OUTF, 1.f);
    k_ln_tanh<<<dim3(NTOT), blk, 0, stream>>>(tbuf, ln2g, ln2b, h16);
    gemm_wmma<128, false, false><<<grd(NTOT, OUTF), blk, 0, stream>>>(
        h16, w2T, out, nullptr, NTOT, OUTF, OUTF, OUTF, OUTF, OUTF, 1.f);
}
